// SPL_11922829214397
// MI455X (gfx1250) — matrix operand order repaired and checked by reading, not compiled, not run on hardware
//
#include <hip/hip_runtime.h>
#include <hip/hip_bf16.h>

typedef __attribute__((ext_vector_type(16))) __bf16 v16bf;
typedef __attribute__((ext_vector_type(8)))  __bf16 v8bf;
typedef __attribute__((ext_vector_type(4)))  __bf16 v4bf;
typedef __attribute__((ext_vector_type(8)))  float  v8f;
typedef __attribute__((ext_vector_type(4)))  float  v4f;
typedef __attribute__((ext_vector_type(4)))  unsigned int u32x4;
typedef __attribute__((ext_vector_type(8)))  int    i32x8;
typedef __attribute__((ext_vector_type(4)))  int    i32x4;

#define NJ      15
#define BATCH   16384
#define CTX     1024
#define HID     64
#define OUTJ    9
#define OUTDIM  (NJ * OUTJ)             // 135
#define MAXCH   34                      // 32 context K-chunks + up to 2 ancestor chunks
#define CHUNK_ELEMS (4 * 32 * 16)       // bf16 elems per K-chunk block = 2048
#define CHUNK_BYTES (CHUNK_ELEMS * 2)   // 4096
#define W0STRIDE (MAXCH * CHUNK_ELEMS)  // bf16 elems per joint in packed W0 = 69632
#define W1STRIDE (2 * 32 * 16)          // 1024

// nanc / ancestor tables (parent-first order, matching reference concat order)
__device__ __constant__ int d_nanc[NJ] = {0,1,2,3,1,2,3,1,2,3,4,3,4,3,4};
__device__ __constant__ int d_anc[NJ][4] = {
  {0,0,0,0},{0,0,0,0},{1,0,0,0},{2,1,0,0},{0,0,0,0},{4,0,0,0},{5,4,0,0},
  {0,0,0,0},{7,0,0,0},{8,7,0,0},{9,8,7,0},{8,7,0,0},{11,8,7,0},{8,7,0,0},{13,8,7,0}};

struct ParamPtrs {
  const float* W0[NJ];
  const float* b0[NJ];
  const float* W1[NJ];
  const float* b1[NJ];
};

__device__ __forceinline__ v16bf cat16(v8bf lo, v8bf hi) {
  v16bf r;
#pragma unroll
  for (int i = 0; i < 8; ++i) { r[i] = lo[i]; r[8 + i] = hi[i]; }
  return r;
}

// TDM: 1-D contiguous copy of nelem8 8-byte elements, global -> LDS.
// D# per CDNA5 ISA: group0 {count=1, lds_addr, global_addr, type=2},
// group1 {data_size=3 (8B), tensor_dim0=nelem8, tile_dim0=nelem8, tile_dim1=0 (1D)}.
// This toolchain's builtin takes 6 args (g0, g1, g2, g3, extra, cpol).
__device__ __forceinline__ void tdm_load_1d(const void* gsrc, unsigned int lds_addr,
                                            unsigned int nelem8) {
  unsigned long long ga = (unsigned long long)gsrc;
  u32x4 g0;
  g0[0] = 1u;                                   // count=1 (user descriptor)
  g0[1] = lds_addr;                             // LDS byte address
  g0[2] = (unsigned int)ga;                     // global_addr[31:0]
  g0[3] = (unsigned int)(ga >> 32) | (2u << 30);// global_addr[56:32] | type=2
  i32x8 g1;
  g1[0] = (int)(3u << 16);                      // workgroup_mask=0, data_size=3 (8B)
  g1[1] = (int)((nelem8 & 0xffffu) << 16);      // tensor_dim0[15:0] in [63:48]
  g1[2] = (int)(nelem8 >> 16);                  // tensor_dim0[31:16] in [79:64]
  g1[3] = (int)(nelem8 << 16);                  // tile_dim0 in [127:112]
  g1[4] = 0;                                    // tile_dim1=0 (1D), tile_dim2=0
  g1[5] = (int)nelem8;                          // tensor_dim0_stride[31:0] (benign for 1D)
  g1[6] = 0;
  g1[7] = 0;
  i32x4 g2 = {0, 0, 0, 0};
  i32x4 g3 = {0, 0, 0, 0};
  i32x8 gx = {0, 0, 0, 0, 0, 0, 0, 0};
  __builtin_amdgcn_tensor_load_to_lds(g0, g1, g2, g3, gx, 0);
}

// ---------------- kernel 1: context fp32 -> bf16 ----------------
__global__ void cvt_kernel(const float* __restrict__ src, __bf16* __restrict__ dst) {
  long i = ((long)blockIdx.x * blockDim.x + threadIdx.x) * 4;
  v4f v = *(const v4f*)(src + i);
  v4bf o;
#pragma unroll
  for (int t = 0; t < 4; ++t) o[t] = (__bf16)v[t];
  *(v4bf*)(dst + i) = o;
}

// ---------------- kernel 2: pack weights into WMMA B layout ----------------
// B k order used here (contiguous, same as A; 32x16): lanes 0-15 hold K=kc*32+0..15 (col = lane&15),
//                           lanes 16-31 hold K=kc*32+16..31.
// Each lane's 16 bf16 stored contiguously -> two b128 loads in main kernel.
__global__ void pack_kernel(ParamPtrs pp,
                            __bf16* __restrict__ w0pack,
                            __bf16* __restrict__ w1pack,
                            float*  __restrict__ bpack) {
  const int bid  = blockIdx.x;
  const int lane = threadIdx.x;        // 32 threads per block
  const int NW0  = NJ * MAXCH * 4;

  if (bid < NW0) {
    int j  = bid / (MAXCH * 4);
    int r  = bid % (MAXCH * 4);
    int kc = r >> 2;
    int t  = r & 3;
    int indim = CTX + 9 * d_nanc[j];
    const float* W = pp.W0[j];
    int n  = t * 16 + (lane & 15);
    int kb = kc * 32 + (lane >> 4) * 16;
    __bf16* dst = w0pack + (long)j * W0STRIDE + (long)((kc * 4 + t) * 32 + lane) * 16;
#pragma unroll
    for (int i = 0; i < 16; ++i) {
      int k = kb + i;
      float v = (k < indim) ? W[(long)k * 64 + n] : 0.0f;
      dst[i] = (__bf16)v;
    }
  } else if (bid < NW0 + NJ * 2) {
    int r  = bid - NW0;
    int j  = r >> 1;
    int kc = r & 1;
    const float* W = pp.W1[j];
    int n  = lane & 15;
    int kb = kc * 32 + (lane >> 4) * 16;
    __bf16* dst = w1pack + (long)j * W1STRIDE + (long)(kc * 32 + lane) * 16;
#pragma unroll
    for (int i = 0; i < 16; ++i) {
      int k = kb + i;
      float v = (k < HID && n < OUTJ) ? W[k * OUTJ + n] : 0.0f;
      dst[i] = (__bf16)v;
    }
  } else {
    int j = bid - (NW0 + NJ * 2);
    for (int t = lane; t < HID; t += 32) bpack[j * 80 + t] = pp.b0[j][t];
    if (lane < 16) bpack[j * 80 + 64 + lane] = (lane < OUTJ) ? pp.b1[j][lane] : 0.0f;
  }
}

// ---------------- kernel 3: fused SPL forward ----------------
// One wave owns 16 batch rows through all 15 joints. Weights (B-operand) are
// shared by all 8 waves of the block: staged per joint into double-buffered LDS
// via TDM (tensor_load_to_lds), prefetching group g+1 while computing group g.
// Per group: phase 1 preloads all 8 A fragments from global (long latency),
// phase 2 streams B from LDS just-in-time (keeps VGPR pressure < 256).
__global__ void __launch_bounds__(256)
spl_main(const __bf16* __restrict__ ctxbf,
         const __bf16* __restrict__ w0pack,
         const __bf16* __restrict__ w1pack,
         const float*  __restrict__ bpack,
         float* __restrict__ outp) {
  __shared__ __bf16 s_preds[8][16 * 160];   // per-wave: 16 rows x preds (stride 160)
  __shared__ __bf16 s_xtra [8][16 * 64];    // per-wave: ancestor-feature K tile
  __shared__ __bf16 s_hid  [8][16 * 64];    // per-wave: hidden activations
  __shared__ __bf16 s_bw   [2][8 * CHUNK_ELEMS]; // block: double-buffered B stage (2x32KB)

  const int lane = threadIdx.x & 31;
  const int wv   = threadIdx.x >> 5;
  const int hi   = lane >> 4;       // 0 or 1
  const int col  = lane & 15;
  const int koff = hi * 16;
  const long rowBase = ((long)blockIdx.x * 8 + wv) * 16;

  __bf16* preds = &s_preds[wv][0];
  __bf16* xtra  = &s_xtra[wv][0];
  __bf16* hid   = &s_hid[wv][0];
  // generic LDS pointer low 32 bits == LDS byte offset (aperture layout)
  const unsigned int bw_lds[2] = {
      (unsigned int)(unsigned long long)(&s_bw[0][0]),
      (unsigned int)(unsigned long long)(&s_bw[1][0])};

  const __bf16* arow = ctxbf + (rowBase + col) * CTX;

  for (int j = 0; j < NJ; ++j) {
    const int na   = d_nanc[j];
    const int etot = 9 * na;
    const int ec   = (etot + 31) >> 5;     // extra K-chunks (0,1,2)
    const __bf16* wb = w0pack + (long)j * W0STRIDE;

    // protect B buffers from overwrite while any wave still reads prev joint
    __syncthreads();

    // prologue: every wave DMAs one 4KB chunk of group 0 into buf0
    tdm_load_1d((const char*)wb + wv * CHUNK_BYTES,
                bw_lds[0] + wv * CHUNK_BYTES, CHUNK_BYTES / 8);

    // stage ancestor predictions as zero-padded [16 x 64] K-extension (overlaps DMA)
    if (na > 0) {
      int anc[4] = {d_anc[j][0], d_anc[j][1], d_anc[j][2], d_anc[j][3]};
      for (int idx = lane; idx < 16 * 64; idx += 32) {
        int r = idx >> 6, c = idx & 63;
        __bf16 v = (__bf16)0.0f;
        if (c < etot) v = preds[r * 160 + anc[c / 9] * 9 + (c % 9)];
        xtra[idx] = v;
      }
    }

    __builtin_amdgcn_s_wait_tensorcnt(0);
    __syncthreads();                        // buf0 (group 0) visible to all waves

    v8f acc0 = {}, acc1 = {}, acc2 = {}, acc3 = {};

    // ---- layer 1: 4 context groups of 8 K-chunks, + optional ancestor group ----
    for (int g = 0; g < 4; ++g) {
      // prefetch next group (context g+1, or ancestor chunks after g==3)
      bool pf = (g < 3) || (ec > 0);
      if (pf) {
        int cig = (g < 3) ? wv : (wv % ec);
        tdm_load_1d((const char*)wb + ((g + 1) * 8 + cig) * CHUNK_BYTES,
                    bw_lds[(g + 1) & 1] + cig * CHUNK_BYTES, CHUNK_BYTES / 8);
      }
      // phase 1: preload all 8 A fragments of this group from global
      const __bf16* ag = arow + g * 256;    // 8 chunks * 32 K
      v16bf aF[8];
#pragma unroll
      for (int kc = 0; kc < 8; ++kc) {
        const __bf16* ap = ag + kc * 32 + koff;
        aF[kc] = cat16(*(const v8bf*)ap, *(const v8bf*)(ap + 8));
      }
      // phase 2: stream B fragments from LDS just-in-time
      const __bf16* bb = &s_bw[g & 1][0];
#pragma unroll
      for (int kc = 0; kc < 8; ++kc) {
        const __bf16* bp = bb + kc * CHUNK_ELEMS + lane * 16;
        v16bf b0 = cat16(*(const v8bf*)(bp + 0 * 512), *(const v8bf*)(bp + 0 * 512 + 8));
        v16bf b1 = cat16(*(const v8bf*)(bp + 1 * 512), *(const v8bf*)(bp + 1 * 512 + 8));
        v16bf b2 = cat16(*(const v8bf*)(bp + 2 * 512), *(const v8bf*)(bp + 2 * 512 + 8));
        v16bf b3 = cat16(*(const v8bf*)(bp + 3 * 512), *(const v8bf*)(bp + 3 * 512 + 8));
        acc0 = __builtin_amdgcn_wmma_f32_16x16x32_bf16(false, aF[kc], false, b0, (short)0, acc0, false, false);
        acc1 = __builtin_amdgcn_wmma_f32_16x16x32_bf16(false, aF[kc], false, b1, (short)0, acc1, false, false);
        acc2 = __builtin_amdgcn_wmma_f32_16x16x32_bf16(false, aF[kc], false, b2, (short)0, acc2, false, false);
        acc3 = __builtin_amdgcn_wmma_f32_16x16x32_bf16(false, aF[kc], false, b3, (short)0, acc3, false, false);
      }
      if (pf) {
        __builtin_amdgcn_s_wait_tensorcnt(0);
        __syncthreads();                    // next group's buffer ready
      }
    }
    // ancestor K-chunks (A from per-wave LDS xtra, B from buf0 = parity of group 4)
    for (int e = 0; e < ec; ++e) {
      const __bf16* ap = xtra + col * 64 + e * 32 + koff;
      v16bf a = cat16(*(const v8bf*)ap, *(const v8bf*)(ap + 8));
      const __bf16* bp = &s_bw[0][0] + e * CHUNK_ELEMS + lane * 16;
      v16bf b0 = cat16(*(const v8bf*)(bp + 0 * 512), *(const v8bf*)(bp + 0 * 512 + 8));
      v16bf b1 = cat16(*(const v8bf*)(bp + 1 * 512), *(const v8bf*)(bp + 1 * 512 + 8));
      v16bf b2 = cat16(*(const v8bf*)(bp + 2 * 512), *(const v8bf*)(bp + 2 * 512 + 8));
      v16bf b3 = cat16(*(const v8bf*)(bp + 3 * 512), *(const v8bf*)(bp + 3 * 512 + 8));
      acc0 = __builtin_amdgcn_wmma_f32_16x16x32_bf16(false, a, false, b0, (short)0, acc0, false, false);
      acc1 = __builtin_amdgcn_wmma_f32_16x16x32_bf16(false, a, false, b1, (short)0, acc1, false, false);
      acc2 = __builtin_amdgcn_wmma_f32_16x16x32_bf16(false, a, false, b2, (short)0, acc2, false, false);
      acc3 = __builtin_amdgcn_wmma_f32_16x16x32_bf16(false, a, false, b3, (short)0, acc3, false, false);
    }

    // bias + ReLU -> bf16 hidden tile in LDS (D elem i: row = i + 8*hi, col = lane&15)
    {
      const float* bj = bpack + j * 80;
      v8f accs[4] = {acc0, acc1, acc2, acc3};
#pragma unroll
      for (int t = 0; t < 4; ++t) {
        float bv = bj[t * 16 + col];
#pragma unroll
        for (int i = 0; i < 8; ++i) {
          float h = accs[t][i] + bv;
          h = fmaxf(h, 0.0f);
          hid[(i + 8 * hi) * 64 + t * 16 + col] = (__bf16)h;
        }
      }
    }

    // ---- layer 2: hid @ W1 (N padded 9->16), K = 64 -> 2 chunks ----
    v8f o = {};
#pragma unroll
    for (int c2 = 0; c2 < 2; ++c2) {
      const __bf16* ap = hid + col * 64 + c2 * 32 + koff;
      v16bf a = cat16(*(const v8bf*)ap, *(const v8bf*)(ap + 8));
      const __bf16* bp = w1pack + (long)j * W1STRIDE + (long)(c2 * 32 + lane) * 16;
      v16bf b = cat16(*(const v8bf*)bp, *(const v8bf*)(bp + 8));
      o = __builtin_amdgcn_wmma_f32_16x16x32_bf16(false, a, false, b, (short)0, o, false, false);
    }

    if (col < OUTJ) {
      float b1v = bpack[j * 80 + 64 + col];
#pragma unroll
      for (int i = 0; i < 8; ++i) {
        int r = i + 8 * hi;
        float v = o[i] + b1v;
        outp[(rowBase + r) * OUTDIM + j * OUTJ + col] = v;
        preds[r * 160 + j * OUTJ + col] = (__bf16)v;
      }
    }
  }
}

extern "C" void kernel_launch(void* const* d_in, const int* in_sizes, int n_in,
                              void* d_out, int out_size, void* d_ws, size_t ws_size,
                              hipStream_t stream) {
  (void)in_sizes; (void)n_in; (void)out_size; (void)ws_size;

  const float* ctx = (const float*)d_in[0];
  ParamPtrs pp;
  for (int j = 0; j < NJ; ++j) {
    pp.W0[j] = (const float*)d_in[1 + 4 * j + 0];
    pp.b0[j] = (const float*)d_in[1 + 4 * j + 1];
    pp.W1[j] = (const float*)d_in[1 + 4 * j + 2];
    pp.b1[j] = (const float*)d_in[1 + 4 * j + 3];
  }

  char* ws = (char*)d_ws;
  size_t off = 0;
  __bf16* ctxbf = (__bf16*)(ws + off); off += (size_t)BATCH * CTX * sizeof(__bf16);
  __bf16* w0p   = (__bf16*)(ws + off); off += (size_t)NJ * W0STRIDE * sizeof(__bf16);
  __bf16* w1p   = (__bf16*)(ws + off); off += (size_t)NJ * W1STRIDE * sizeof(__bf16);
  float*  bp    = (float*) (ws + off); off += (size_t)NJ * 80 * sizeof(float);
  float*  outp  = (float*)d_out;

  // 1) context -> bf16 (re-read once per joint thereafter)
  cvt_kernel<<<(BATCH * CTX) / (256 * 4), 256, 0, stream>>>(ctx, ctxbf);
  // 2) swizzle weights into WMMA B-operand layout (zero-padded)
  pack_kernel<<<NJ * MAXCH * 4 + NJ * 2 + NJ, 32, 0, stream>>>(pp, w0p, w1p, bp);
  // 3) fused forward: 1024 waves, each owns 16 rows through all 15 joints
  spl_main<<<BATCH / (16 * 8), 256, 0, stream>>>(ctxbf, w0p, w1p, bp, outp);
}
